// Exercise_KC_GraphConvolution_41996190220783
// MI455X (gfx1250) — matrix operand order repaired and checked by reading, not compiled, not run on hardware
//
#include <hip/hip_runtime.h>
#include <stdint.h>

#define E_ROWS   50000
#define IN_F     512
#define OUT_F    512
#define KC_ROWS  2048

#define KSTEP      32    // K-panel width
#define A_STRIDE   40    // staged A panel row stride in bf16 (32 + 8 pad)
#define B_STRIDE   40    // staged B panel column stride in bf16 (32 + 8 pad)
#define EXW_STRIDE 520   // exWh LDS row stride in bf16 (512 + 8 pad)

typedef __bf16 bf16_t;
typedef __attribute__((ext_vector_type(16))) __bf16 v16bf;
typedef __attribute__((ext_vector_type(8)))  __bf16 v8bf;
typedef __attribute__((ext_vector_type(4)))  __bf16 v4bf;
typedef __attribute__((ext_vector_type(8)))  float   v8f;

__device__ __forceinline__ v16bf concat8(v8bf lo, v8bf hi) {
  return __builtin_shufflevector(lo, hi, 0,1,2,3,4,5,6,7,8,9,10,11,12,13,14,15);
}

__device__ __forceinline__ v8f zero8() {
  v8f z = {0.f,0.f,0.f,0.f,0.f,0.f,0.f,0.f};
  return z;
}

__device__ __forceinline__ v8f wmma_bf16(v16bf a, v16bf b, v8f c) {
  // D = A(16x32 bf16) x B(32x16 bf16) + C(16x16 f32)
  return __builtin_amdgcn_wmma_f32_16x16x32_bf16(false, a, false, b, (short)0, c, false, false);
}

// A fragment (16x32 bf16) from a row-major bf16 tile (contiguous K: lane half h holds K=16h..16h+15).
__device__ __forceinline__ v16bf load_a_frag(const bf16_t* __restrict__ tile,
                                             int stride, int k_local, int lane) {
  const int half = lane >> 4;
  const int r    = lane & 15;
  const bf16_t* p = tile + r * stride + k_local + 16 * half;
  v8bf c0 = *(const v8bf*)(p);
  v8bf c1 = *(const v8bf*)(p + 8);
  return concat8(c0, c1);
}

// B fragment (32x16) from an LDS panel: 512 columns x 32 K, column stride B_STRIDE.
// lanes 0-15: col=n+l, K=[0..15]; lanes 16-31: col=n+l, K=[16..31]
__device__ __forceinline__ v16bf load_b_frag_lds(const bf16_t* __restrict__ panel,
                                                 int n, int lane) {
  const int half = lane >> 4;
  const int col  = n + (lane & 15);
  const bf16_t* p = panel + col * B_STRIDE + 16 * half;
  v8bf c0 = *(const v8bf*)(p);
  v8bf c1 = *(const v8bf*)(p + 8);
  return concat8(c0, c1);
}

// B fragment straight from a column-major bf16 matrix in global memory (L2-hot).
__device__ __forceinline__ v16bf load_b_frag(const bf16_t* __restrict__ bcm,
                                             long ldb, int n, int k0, int lane) {
  const int half = lane >> 4;
  const int col  = n + (lane & 15);
  const bf16_t* p = bcm + (long)col * ldb + k0 + 16 * half;
  v8bf c0 = *(const v8bf*)(p);
  v8bf c1 = *(const v8bf*)(p + 8);
  return concat8(c0, c1);
}

// 256 threads stage a 32-row x 32-col f32 panel as bf16 into LDS (A operand).
__device__ __forceinline__ void stageA32(bf16_t* __restrict__ dst,
                                         const float* __restrict__ src,
                                         long ld, int maxr, int tid) {
  const int row  = tid >> 3;
  const int col  = (tid & 7) << 2;
  const int crow = row <= maxr ? row : maxr;   // clamp for ragged last M-tile
  const float4 v = *(const float4*)(src + (long)crow * ld + col);
  v4bf b = { (bf16_t)v.x, (bf16_t)v.y, (bf16_t)v.z, (bf16_t)v.w };
  *(v4bf*)(dst + row * A_STRIDE + col) = b;
}

// Async-stage a 512-col x 32-K bf16 B panel into LDS via the CDNA5 ASYNC path.
// Each thread moves 2 columns (4x b128 each); tracked by ASYNCcnt.
__device__ __forceinline__ void stageB32_async(bf16_t* __restrict__ dstB,
                                               const bf16_t* __restrict__ Bcm,
                                               long ldb, int k0, int tid) {
#pragma unroll
  for (int cc = 0; cc < 2; ++cc) {
    const int c = tid * 2 + cc;
    // generic LDS pointer: low 32 bits are the LDS byte address (ISA 10.2)
    const uint32_t lds0 = (uint32_t)(uintptr_t)(dstB + c * B_STRIDE);
    const uint32_t voff = (uint32_t)(((long)c * ldb + k0) * 2);
#pragma unroll
    for (int q = 0; q < 4; ++q) {
      asm volatile("global_load_async_to_lds_b128 %0, %1, %2"
                   :: "v"(lds0 + q * 16u), "v"(voff + q * 16u), "s"(Bcm)
                   : "memory");
    }
  }
}

__device__ __forceinline__ void wait_async0() {
  asm volatile("s_wait_asynccnt 0" ::: "memory");
}

// Double-buffered tile-GEMM accumulate:
//   acc[8] += bf16(srcA[32 x ktot]) @ Bcm[:, nbase..nbase+128]
// A staged via VGPR convert path, B staged via async LDS copies; one barrier per panel.
__device__ __forceinline__ void gemm_accum(v8f acc[8], int ktot,
                                           const float* __restrict__ srcA, long lda,
                                           const bf16_t* __restrict__ Bcm, long ldb,
                                           bf16_t* __restrict__ sA2,   // 2 * 32*A_STRIDE
                                           bf16_t* __restrict__ sB2,   // 2 * 512*B_STRIDE
                                           int maxr, int tid, int lane,
                                           int m_off, int nbase) {
  const int iters = ktot / KSTEP;
  stageA32(sA2, srcA, lda, maxr, tid);
  stageB32_async(sB2, Bcm, ldb, 0, tid);
  wait_async0();
  __syncthreads();
  int cur = 0;
  for (int it = 0; it < iters; ++it) {
    const int k0 = it * KSTEP;
    if (it + 1 < iters) {
      stageA32(sA2 + (cur ^ 1) * (32 * A_STRIDE), srcA + k0 + KSTEP, lda, maxr, tid);
      stageB32_async(sB2 + (cur ^ 1) * (512 * B_STRIDE), Bcm, ldb, k0 + KSTEP, tid);
      if (it + 2 < iters) {
        // prime L2 two panels ahead (HBM stream for adj)
        const int row  = tid >> 3;
        const int col  = (tid & 7) << 2;
        const int crow = row <= maxr ? row : maxr;
        __builtin_prefetch((const void*)(srcA + (long)crow * lda + k0 + 2 * KSTEP + col), 0, 1);
      }
    }
    const bf16_t* tileA = sA2 + cur * (32 * A_STRIDE) + m_off * A_STRIDE;
    const bf16_t* tileB = sB2 + cur * (512 * B_STRIDE);
    v16bf af = load_a_frag(tileA, A_STRIDE, 0, lane);
#pragma unroll
    for (int jg = 0; jg < 8; jg += 4) {
      v16bf b0 = load_b_frag_lds(tileB, nbase + (jg + 0) * 16, lane);
      v16bf b1 = load_b_frag_lds(tileB, nbase + (jg + 1) * 16, lane);
      v16bf b2 = load_b_frag_lds(tileB, nbase + (jg + 2) * 16, lane);
      v16bf b3 = load_b_frag_lds(tileB, nbase + (jg + 3) * 16, lane);
      acc[jg + 0] = wmma_bf16(af, b0, acc[jg + 0]);
      acc[jg + 1] = wmma_bf16(af, b1, acc[jg + 1]);
      acc[jg + 2] = wmma_bf16(af, b2, acc[jg + 2]);
      acc[jg + 3] = wmma_bf16(af, b3, acc[jg + 3]);
    }
    wait_async0();     // our next-panel async copies must land before we signal
    __syncthreads();
    cur ^= 1;
  }
}

// ---------------------------------------------------------------------------
// Kernel 0: W1, W2 (f32 row-major [K=IN_F, N=OUT_F]) -> bf16 column-major
// ---------------------------------------------------------------------------
__global__ void convert_w_kernel(const float* __restrict__ W1,
                                 const float* __restrict__ W2,
                                 bf16_t* __restrict__ W1cm,
                                 bf16_t* __restrict__ W2cm) {
  const int idx = blockIdx.x * blockDim.x + threadIdx.x;
  if (idx < IN_F * OUT_F) {
    const int k = idx % IN_F;     // contiguous write along K
    const int n = idx / IN_F;
    W1cm[idx] = (bf16_t)W1[(long)k * OUT_F + n];
    W2cm[idx] = (bf16_t)W2[(long)k * OUT_F + n];
  }
}

// ---------------------------------------------------------------------------
// Kernel 1: kcWb[n*KC_ROWS + m] = bf16( (kc_h @ W1)[m][n] )  (column-major)
// ---------------------------------------------------------------------------
__global__ __launch_bounds__(256) void kc_gemm_kernel(const float* __restrict__ kch,
                                                      const bf16_t* __restrict__ W1cm,
                                                      bf16_t* __restrict__ kcWb) {
  __shared__ bf16_t sA[2 * 32 * A_STRIDE];
  __shared__ bf16_t sB[2 * 512 * B_STRIDE];

  const int tid   = threadIdx.x;
  const int lane  = tid & 31;
  const int wid   = tid >> 5;
  const int half  = lane >> 4;
  const int l15   = lane & 15;
  const int m_off = (wid & 1) * 16;
  const int nbase = (wid >> 1) * 128;
  const int m0    = blockIdx.x * 32;

  v8f acc[8];
#pragma unroll
  for (int j = 0; j < 8; ++j) acc[j] = zero8();

  gemm_accum(acc, IN_F, kch + (long)m0 * IN_F, IN_F, W1cm, IN_F,
             sA, sB, 31, tid, lane, m_off, nbase);

  // store bf16 column-major: kcWb[col * KC_ROWS + row]
#pragma unroll
  for (int j = 0; j < 8; ++j) {
#pragma unroll
    for (int i = 0; i < 8; ++i) {
      const int gm = m0 + m_off + half * 8 + i;
      const int gn = nbase + j * 16 + l15;
      kcWb[(long)gn * KC_ROWS + gm] = (bf16_t)acc[j][i];
    }
  }
}

// ---------------------------------------------------------------------------
// Kernel 2 (fused): per 32 exercise rows:
//   exWh = exh @ W1  (WMMA -> bf16 in LDS) ; acc2 = exWh @ W2 (WMMA, A from LDS)
//   acc  = adj @ kc_Wh (WMMA, K=2048, async-staged B panels)
//   out  = elu(acc * acc2)
// ---------------------------------------------------------------------------
__global__ __launch_bounds__(256) void fused_gcn_kernel(const float* __restrict__ exh,
                                                        const float* __restrict__ adj,
                                                        const bf16_t* __restrict__ W1cm,
                                                        const bf16_t* __restrict__ W2cm,
                                                        const bf16_t* __restrict__ kcWb,
                                                        float* __restrict__ out) {
  __shared__ bf16_t sA[2 * 32 * A_STRIDE];
  __shared__ bf16_t sB[2 * 512 * B_STRIDE];
  __shared__ bf16_t sExW[32 * EXW_STRIDE];

  const int tid   = threadIdx.x;
  const int lane  = tid & 31;
  const int wid   = tid >> 5;
  const int half  = lane >> 4;
  const int l15   = lane & 15;
  const int m_off = (wid & 1) * 16;       // 2 M-groups of 16 rows
  const int nbase = (wid >> 1) * 128;     // 4 N-groups of 128 cols
  const int m0    = blockIdx.x * 32;
  const int rem   = E_ROWS - 1 - m0;
  const int maxr  = rem < 31 ? rem : 31;

  // ---- Stage 1: exWh tile -> LDS (bf16) --------------------------------
  {
    v8f acc[8];
#pragma unroll
    for (int j = 0; j < 8; ++j) acc[j] = zero8();

    gemm_accum(acc, IN_F, exh + (long)m0 * IN_F, IN_F, W1cm, IN_F,
               sA, sB, maxr, tid, lane, m_off, nbase);

#pragma unroll
    for (int j = 0; j < 8; ++j) {
#pragma unroll
      for (int i = 0; i < 8; ++i) {
        const int r = m_off + half * 8 + i;
        const int c = nbase + j * 16 + l15;
        sExW[r * EXW_STRIDE + c] = (bf16_t)acc[j][i];
      }
    }
  }
  __syncthreads();

  // ---- Stage 3: acc2 = exWh @ W2 (A from LDS, B from L2-hot W2cm) ------
  v8f acc2[8];
#pragma unroll
  for (int j = 0; j < 8; ++j) acc2[j] = zero8();

  for (int k0 = 0; k0 < OUT_F; k0 += 32) {
    v16bf af = load_a_frag(sExW + m_off * EXW_STRIDE, EXW_STRIDE, k0, lane);
#pragma unroll
    for (int jg = 0; jg < 8; jg += 4) {
      v16bf b0 = load_b_frag(W2cm, OUT_F, nbase + (jg + 0) * 16, k0, lane);
      v16bf b1 = load_b_frag(W2cm, OUT_F, nbase + (jg + 1) * 16, k0, lane);
      v16bf b2 = load_b_frag(W2cm, OUT_F, nbase + (jg + 2) * 16, k0, lane);
      v16bf b3 = load_b_frag(W2cm, OUT_F, nbase + (jg + 3) * 16, k0, lane);
      acc2[jg + 0] = wmma_bf16(af, b0, acc2[jg + 0]);
      acc2[jg + 1] = wmma_bf16(af, b1, acc2[jg + 1]);
      acc2[jg + 2] = wmma_bf16(af, b2, acc2[jg + 2]);
      acc2[jg + 3] = wmma_bf16(af, b3, acc2[jg + 3]);
    }
  }
  __syncthreads();

  // ---- Stage 2: acc = adj @ kc_Wh (K = 2048, async B panels) -----------
  v8f acc[8];
#pragma unroll
  for (int j = 0; j < 8; ++j) acc[j] = zero8();

  gemm_accum(acc, KC_ROWS, adj + (long)m0 * KC_ROWS, KC_ROWS, kcWb, KC_ROWS,
             sA, sB, maxr, tid, lane, m_off, nbase);

  // ---- combine + ELU + store ------------------------------------------
#pragma unroll
  for (int j = 0; j < 8; ++j) {
#pragma unroll
    for (int i = 0; i < 8; ++i) {
      const int gm = m0 + m_off + half * 8 + i;
      if (gm < E_ROWS) {
        const float v = acc[j][i] * acc2[j][i];
        const float r = v > 0.f ? v : (__expf(v) - 1.f);
        out[(long)gm * OUT_F + nbase + j * 16 + l15] = r;
      }
    }
  }
}

// ---------------------------------------------------------------------------
extern "C" void kernel_launch(void* const* d_in, const int* in_sizes, int n_in,
                              void* d_out, int out_size, void* d_ws, size_t ws_size,
                              hipStream_t stream) {
  const float* exh = (const float*)d_in[0];   // [50000, 512]
  const float* kch = (const float*)d_in[1];   // [2048, 512]
  const float* adj = (const float*)d_in[2];   // [50000, 2048]
  const float* W1  = (const float*)d_in[3];   // [512, 512]
  const float* W2  = (const float*)d_in[4];   // [512, 512]
  float* out = (float*)d_out;                 // [50000, 512]

  // workspace: bf16 column-major weights + kc_Wh  (~3 MB total)
  bf16_t* W1cm = (bf16_t*)d_ws;                  // 512*512
  bf16_t* W2cm = W1cm + (long)IN_F * OUT_F;      // 512*512
  bf16_t* kcWb = W2cm + (long)IN_F * OUT_F;      // 2048*512

  convert_w_kernel<<<(IN_F * OUT_F + 255) / 256, 256, 0, stream>>>(W1, W2, W1cm, W2cm);
  kc_gemm_kernel<<<KC_ROWS / 32, 256, 0, stream>>>(kch, W1cm, kcWb);
  fused_gcn_kernel<<<(E_ROWS + 31) / 32, 256, 0, stream>>>(exh, adj, W1cm, W2cm, kcWb, out);
}
